// GruPredictor_44530220925411
// MI455X (gfx1250) — matrix operand order repaired and checked by reading, not compiled, not run on hardware
//
#include <hip/hip_runtime.h>
#include <math.h>

#define STATE_DIM 6
#define HIDDEN    128
#define GATES     384      // 3 * HIDDEN
#define SEQ       128
#define FCL       8
#define TAIL      4        // (FCL+1)/2 ... = 4 for FCL=8
#define LCOMB     12       // TAIL + FCL
#define BTILE     16       // batch rows per block (WMMA M)
#define THREADS   128      // 4 waves of 32
#define NT_PER_WAVE 6      // 24 N-tiles (384/16) split over 4 waves

typedef __attribute__((ext_vector_type(16))) _Float16 v16h;
typedef __attribute__((ext_vector_type(8)))  _Float16 v8h;
typedef __attribute__((ext_vector_type(8)))  float    v8f;

struct __align__(32) Smem {
    _Float16 w16[GATES * HIDDEN];      // 98304 B : w_hh in f16, row-major [384][128]
    float    wih[GATES * STATE_DIM];   //  9216 B : w_ih f32
    float    bih[GATES];               //  1536 B
    float    bhh[GATES];               //  1536 B
    float    gh [BTILE * GATES];       // 24576 B : h @ w_hh.T accumulators (f32)
    float    h32[BTILE * HIDDEN];      //  8192 B : hidden state f32
    _Float16 h16[BTILE * HIDDEN];      //  4096 B : hidden state f16 (WMMA A source)
    float    xs [BTILE * STATE_DIM];   //   384 B : current cell input (x_t or st)
    float    scores[BTILE * LCOMB];    //   768 B : attention logits
    float    states[BTILE * LCOMB * STATE_DIM]; // 4608 B : state_comb
};                                     // total ~153 KB < 320 KB WGP LDS

__device__ __forceinline__ float sigmoidf_(float v) { return 1.0f / (1.0f + __expf(-v)); }

// WMMA stage: gh[16,384] = h16[16,128] @ w_hh.T (f16 inputs, f32 accumulate).
// Each wave computes a 96-column slice: 6 N-tiles x 4 K-chunks of v_wmma_f32_16x16x32_f16.
__device__ __forceinline__ void gh_wmma(Smem* s, int tid) {
    const int wave = tid >> 5;
    const int lane = tid & 31;
    const int mrow = lane & 15;
    const int hi   = lane >> 4;   // lane half-select

    // A fragments: contiguous k order (this code's choice, matched by B).
    // lanes 0-15: row M=lane, K in {k0..k0+15}; lanes 16-31 shifted by 16.
    v16h a[4];
#pragma unroll
    for (int c = 0; c < 4; ++c) {
        const int k0 = 32 * c + hi * 16;
        const v8h lo = *(const v8h*)(&s->h16[mrow * HIDDEN + k0]);
        const v8h up = *(const v8h*)(&s->h16[mrow * HIDDEN + k0 + 8]);
#pragma unroll
        for (int i = 0; i < 8; ++i) { a[c][i] = lo[i]; a[c][i + 8] = up[i]; }
    }

#pragma unroll
    for (int nt = 0; nt < NT_PER_WAVE; ++nt) {
        const int nbase = (wave * NT_PER_WAVE + nt) * 16;
        const int ncol  = nbase + mrow;   // output gate column owned by this lane
        v8f acc = {};
#pragma unroll
        for (int c = 0; c < 4; ++c) {
            // B fragment: same contiguous k order as A. Lane (N=mrow) holds 16 consecutive K
            // values of column N => 16 contiguous halves of w_hh row `ncol`.
            const int kb = 32 * c + hi * 16;
            const v16h b = *(const v16h*)(&s->w16[ncol * HIDDEN + kb]);
            acc = __builtin_amdgcn_wmma_f32_16x16x32_f16(
                false, a[c], false, b, (short)0, acc, false, false);
        }
        // C/D layout: reg r -> (M = r + 8*hi, N = mrow)
#pragma unroll
        for (int r = 0; r < 8; ++r)
            s->gh[(r + hi * 8) * GATES + nbase + mrow] = acc[r];
    }
}

// Gate math + hidden-state update (includes the tiny 6-wide input GEMM).
__device__ __forceinline__ void gate_update(Smem* s, int tid) {
    for (int idx = tid; idx < BTILE * HIDDEN; idx += THREADS) {
        const int m = idx >> 7;
        const int j = idx & (HIDDEN - 1);
        float ir = s->bih[j];
        float iz = s->bih[j + HIDDEN];
        float in = s->bih[j + 2 * HIDDEN];
#pragma unroll
        for (int k = 0; k < STATE_DIM; ++k) {
            const float xv = s->xs[m * STATE_DIM + k];
            ir += xv * s->wih[j * STATE_DIM + k];
            iz += xv * s->wih[(j + HIDDEN) * STATE_DIM + k];
            in += xv * s->wih[(j + 2 * HIDDEN) * STATE_DIM + k];
        }
        const float hr = s->gh[m * GATES + j]              + s->bhh[j];
        const float hz = s->gh[m * GATES + j + HIDDEN]     + s->bhh[j + HIDDEN];
        const float hn = s->gh[m * GATES + j + 2 * HIDDEN] + s->bhh[j + 2 * HIDDEN];
        const float r = sigmoidf_(ir + hr);
        const float z = sigmoidf_(iz + hz);
        const float n = tanhf(in + r * hn);
        const float hnew = (1.0f - z) * n + z * s->h32[idx];
        s->h32[idx] = hnew;
        s->h16[idx] = (_Float16)hnew;
    }
}

__device__ __forceinline__ void att_score(Smem* s, int tid, int slot,
                                          const float* __restrict__ w_att, float b_att) {
    if (tid < BTILE) {
        float acc = b_att;
        for (int k = 0; k < HIDDEN; ++k) acc += s->h32[tid * HIDDEN + k] * w_att[k];
        s->scores[tid * LCOMB + slot] = acc;
    }
}

__device__ __forceinline__ void fc_step(Smem* s, int tid,
                                        const float* __restrict__ w_fc,
                                        const float* __restrict__ b_fc) {
    if (tid < BTILE * STATE_DIM) {
        const int m = tid / STATE_DIM;
        const int o = tid % STATE_DIM;
        float acc = b_fc[o];
        for (int k = 0; k < HIDDEN; ++k) acc += s->h32[m * HIDDEN + k] * w_fc[o * HIDDEN + k];
        s->xs[tid] = acc;
    }
}

__global__ __launch_bounds__(THREADS)
void gru_predictor_kernel(const float* __restrict__ state_seq,
                          const float* __restrict__ w_ih,
                          const float* __restrict__ w_hh,
                          const float* __restrict__ b_ih,
                          const float* __restrict__ b_hh,
                          const float* __restrict__ w_fc,
                          const float* __restrict__ b_fc,
                          const float* __restrict__ w_att,
                          const float* __restrict__ b_att,
                          float* __restrict__ out) {
    __shared__ Smem s;
    const int tid = threadIdx.x;
    const int b0  = blockIdx.x * BTILE;

    // Stage weights into LDS (resident for all 136 steps); zero the hidden state.
    for (int i = tid; i < GATES * HIDDEN; i += THREADS) s.w16[i] = (_Float16)w_hh[i];
    for (int i = tid; i < GATES * STATE_DIM; i += THREADS) s.wih[i] = w_ih[i];
    for (int i = tid; i < GATES; i += THREADS) { s.bih[i] = b_ih[i]; s.bhh[i] = b_hh[i]; }
    for (int i = tid; i < BTILE * HIDDEN; i += THREADS) {
        s.h32[i] = 0.0f; s.h16[i] = (_Float16)0.0f;
    }
    __syncthreads();

    // ---- GRU encoder over SEQ steps ----
    for (int t = 0; t < SEQ; ++t) {
        if (tid < BTILE * STATE_DIM) {
            const int m = tid / STATE_DIM, k = tid % STATE_DIM;
            s.xs[tid] = state_seq[((size_t)(b0 + m) * SEQ + t) * STATE_DIM + k];
        }
        __syncthreads();
        gh_wmma(&s, tid);
        __syncthreads();
        gate_update(&s, tid);
        __syncthreads();
        if (t >= SEQ - TAIL) att_score(&s, tid, t - (SEQ - TAIL), w_att, b_att[0]);
    }
    __syncthreads();

    // state_comb slots 0..TAIL-1 = last TAIL input states
    if (tid < BTILE * STATE_DIM) {
        const int m = tid / STATE_DIM, k = tid % STATE_DIM;
#pragma unroll
        for (int l = 0; l < TAIL; ++l)
            s.states[(m * LCOMB + l) * STATE_DIM + k] =
                state_seq[((size_t)(b0 + m) * SEQ + (SEQ - TAIL + l)) * STATE_DIM + k];
    }
    // st0 = fc(h_n)
    fc_step(&s, tid, w_fc, b_fc);
    __syncthreads();

    // ---- Autoregressive rollout (records (st, h) BEFORE each update) ----
    for (int f = 0; f < FCL; ++f) {
        if (tid < BTILE * STATE_DIM) {
            const int m = tid / STATE_DIM, k = tid % STATE_DIM;
            s.states[(m * LCOMB + TAIL + f) * STATE_DIM + k] = s.xs[tid];
        }
        att_score(&s, tid, TAIL + f, w_att, b_att[0]);
        __syncthreads();
        gh_wmma(&s, tid);          // x for this step is the current st in s.xs
        __syncthreads();
        gate_update(&s, tid);
        __syncthreads();
        fc_step(&s, tid, w_fc, b_fc);  // st_{f+1}
        __syncthreads();
    }

    // ---- Softmax over the 12 combined steps + weighted state sum ----
    if (tid < BTILE) {
        float mx = -1e30f;
#pragma unroll
        for (int l = 0; l < LCOMB; ++l) mx = fmaxf(mx, s.scores[tid * LCOMB + l]);
        float w[LCOMB];
        float wsum = 0.0f;
#pragma unroll
        for (int l = 0; l < LCOMB; ++l) {
            w[l] = __expf(s.scores[tid * LCOMB + l] - mx);
            wsum += w[l];
        }
        const float inv = 1.0f / wsum;
#pragma unroll
        for (int d = 0; d < STATE_DIM; ++d) {
            float acc = 0.0f;
#pragma unroll
            for (int l = 0; l < LCOMB; ++l)
                acc += w[l] * s.states[(tid * LCOMB + l) * STATE_DIM + d];
            out[(size_t)(b0 + tid) * STATE_DIM + d] = acc * inv;
        }
    }
}

extern "C" void kernel_launch(void* const* d_in, const int* in_sizes, int n_in,
                              void* d_out, int out_size, void* d_ws, size_t ws_size,
                              hipStream_t stream) {
    const float* state_seq = (const float*)d_in[0];
    const float* w_ih      = (const float*)d_in[1];
    const float* w_hh      = (const float*)d_in[2];
    const float* b_ih      = (const float*)d_in[3];
    const float* b_hh      = (const float*)d_in[4];
    const float* w_fc      = (const float*)d_in[5];
    const float* b_fc      = (const float*)d_in[6];
    const float* w_att     = (const float*)d_in[7];
    const float* b_att     = (const float*)d_in[8];
    float* out             = (float*)d_out;

    const int batch = in_sizes[0] / (SEQ * STATE_DIM);   // 8192
    dim3 grid(batch / BTILE);                            // 512 blocks
    gru_predictor_kernel<<<grid, THREADS, 0, stream>>>(
        state_seq, w_ih, w_hh, b_ih, b_hh, w_fc, b_fc, w_att, b_att, out);
}
